// Transducer_55740085567624
// MI455X (gfx1250) — matrix operand order repaired and checked by reading, not compiled, not run on hardware
//
#include <hip/hip_runtime.h>
#include <hip/hip_bf16.h>

typedef __attribute__((ext_vector_type(16))) __bf16 v16bf;
typedef __attribute__((ext_vector_type(8)))  __bf16 v8bf;
typedef __attribute__((ext_vector_type(8)))  float  v8f;

#define B_   4
#define T_   250
#define U_   120
#define U1_  121
#define V_   512
#define D_   512
#define H_   512
#define NEGV (-1e30f)

#define TOTALP (B_ * T_ * U1_)     // 121000 joint positions
#define MTILE  64                  // rows (positions) per block
#define NBLK   ((TOTALP + MTILE - 1) / MTILE)   // 1891
#define SA_STRIDE 520              // bf16 halves per A row (pad vs banks)
#define SL_STRIDE 516              // floats per logits row
#define SMEM_BYTES (MTILE * SA_STRIDE * 2)      // 66560 B (> 64KB -> dynamic LDS)

// ---------------------------------------------------------------------------
// Kernel 1: W_outT[v][h] = (bf16) W_out[h][v]   (512x512, one-shot, L2-resident)
// ---------------------------------------------------------------------------
__global__ void k_wt(const float* __restrict__ W_out, __bf16* __restrict__ WT) {
  int i = blockIdx.x * 256 + threadIdx.x;     // i in [0, V*H)
  int v = i >> 9, h = i & 511;
  WT[v * H_ + h] = (__bf16)W_out[h * V_ + v];
}

// ---------------------------------------------------------------------------
// Kernel 2: enc_p[row][h] = encoder_out[row][:] . W_enc[:][h]   (fp32, small)
// ---------------------------------------------------------------------------
__global__ __launch_bounds__(256)
void k_encp(const float* __restrict__ x, const float* __restrict__ W,
            float* __restrict__ out) {
  __shared__ float xs[D_];
  const float* xr = x + (size_t)blockIdx.x * D_;
  for (int d = threadIdx.x; d < D_; d += 256) xs[d] = xr[d];
  __syncthreads();
  int h0 = threadIdx.x, h1 = threadIdx.x + 256;
  float a0 = 0.f, a1 = 0.f;
  for (int d = 0; d < D_; ++d) {
    float xv = xs[d];
    a0 += xv * W[d * H_ + h0];
    a1 += xv * W[d * H_ + h1];
  }
  out[(size_t)blockIdx.x * H_ + h0] = a0;
  out[(size_t)blockIdx.x * H_ + h1] = a1;
}

// ---------------------------------------------------------------------------
// Kernel 3: dec_p[b][u][h] = emb[dec_in[b][u]][:] . W_dec[:][h]
// ---------------------------------------------------------------------------
__global__ __launch_bounds__(256)
void k_decp(const int* __restrict__ labels, const float* __restrict__ emb,
            const float* __restrict__ W, float* __restrict__ out) {
  int b = blockIdx.x / U1_, u = blockIdx.x % U1_;
  int idx = (u == 0) ? 0 : labels[b * U_ + (u - 1)];
  __shared__ float xs[D_];
  const float* xr = emb + (size_t)idx * D_;
  for (int d = threadIdx.x; d < D_; d += 256) xs[d] = xr[d];
  __syncthreads();
  int h0 = threadIdx.x, h1 = threadIdx.x + 256;
  float a0 = 0.f, a1 = 0.f;
  for (int d = 0; d < D_; ++d) {
    float xv = xs[d];
    a0 += xv * W[d * H_ + h0];
    a1 += xv * W[d * H_ + h1];
  }
  out[(size_t)blockIdx.x * H_ + h0] = a0;
  out[(size_t)blockIdx.x * H_ + h1] = a1;
}

// ---------------------------------------------------------------------------
// Kernel 4 (dominant, WMMA): per block = 64 flat joint positions.
//   h-tile(64x512) = tanh(enc_p + dec_p + b_joint) -> bf16 in dynamic LDS
//   logits(64x512) = h-tile @ W_outT^T via v_wmma_f32_16x16x32_bf16
//       wave w: 4 M-tiles x 4 N-tiles (cols [w*64, w*64+64)), K=512
//       B frags register double-buffered (prefetch k+1 before k's WMMAs)
//   epilogue: four 16-row LSE passes reusing the A-tile LDS (aliased)
// ---------------------------------------------------------------------------
__global__ __launch_bounds__(256)
void k_joint(const float* __restrict__ enc_p, const float* __restrict__ dec_p,
             const float* __restrict__ b_joint, const __bf16* __restrict__ WT,
             const float* __restrict__ b_out, const int* __restrict__ labels,
             float* __restrict__ blank_lp, float* __restrict__ label_lp) {
  extern __shared__ __align__(16) char smem[];  // SMEM_BYTES
  __bf16* sA = (__bf16*)smem;                 // 64 x 520 halves (GEMM phase)
  float*  sL = (float*)smem;                  // 16 x 516 floats (epilogue, aliased)

  const int p0 = blockIdx.x * MTILE;

  // ---- phase 1: build bf16 A tile (64 x 512) ----
  for (int e = threadIdx.x; e < MTILE * H_; e += 256) {
    int m = e >> 9, h = e & 511;
    int p = p0 + m;
    float val = 0.f;
    if (p < TOTALP) {
      int b   = p / (T_ * U1_);
      int rem = p - b * (T_ * U1_);
      int t   = rem / U1_;
      int u   = rem - t * U1_;
      val = tanhf(enc_p[(size_t)(b * T_ + t) * H_ + h] +
                  dec_p[(size_t)(b * U1_ + u) * H_ + h] + b_joint[h]);
    }
    sA[m * SA_STRIDE + h] = (__bf16)val;
  }
  __syncthreads();

  // ---- phase 2: WMMA GEMM, K=512 in 16 steps of 32, pipelined B frags ----
  int lane = threadIdx.x & 31;
  int w    = threadIdx.x >> 5;
  int row  = lane & 15;          // A row-in-tile / B column-in-tile / C column
  int hi   = lane >> 4;          // lane half select
  const int n0 = w * 64;

  const __bf16* bptr[4];
#pragma unroll
  for (int nt = 0; nt < 4; ++nt)
    bptr[nt] = WT + (size_t)(n0 + nt * 16 + row) * H_ + hi * 16;

  v8f   acc[4][4] = {};
  v16bf bfrag[2][4];
#pragma unroll
  for (int nt = 0; nt < 4; ++nt)           // prologue: k=0 B fragments
    bfrag[0][nt] = *(const v16bf*)(bptr[nt]);

#pragma unroll
  for (int k = 0; k < 16; ++k) {
    const int cur = k & 1, nxt = cur ^ 1;
    if (k < 15) {                          // prefetch k+1 B fragments
#pragma unroll
      for (int nt = 0; nt < 4; ++nt)
        bfrag[nxt][nt] = *(const v16bf*)(bptr[nt] + (k + 1) * 32);
    }
    v16bf afrag[4];
#pragma unroll
    for (int mt = 0; mt < 4; ++mt) {
      int abase = (mt * 16 + row) * SA_STRIDE + k * 32 + hi * 16;
      v8bf a_lo = *(const v8bf*)(sA + abase);
      v8bf a_hi = *(const v8bf*)(sA + abase + 8);
      afrag[mt] = __builtin_shufflevector(a_lo, a_hi,
          0, 1, 2, 3, 4, 5, 6, 7, 8, 9, 10, 11, 12, 13, 14, 15);
    }
#pragma unroll
    for (int mt = 0; mt < 4; ++mt)
#pragma unroll
      for (int nt = 0; nt < 4; ++nt)
        acc[mt][nt] = __builtin_amdgcn_wmma_f32_16x16x32_bf16(
            false, afrag[mt], false, bfrag[cur][nt], (short)0, acc[mt][nt],
            false, false);
  }
  __syncthreads();    // A tile dead; sL may now alias it

  // ---- phase 3: four 16-row epilogue passes (spill + LSE + emit) ----
#pragma unroll
  for (int mt = 0; mt < 4; ++mt) {
#pragma unroll
    for (int nt = 0; nt < 4; ++nt) {
      int col = n0 + nt * 16 + row;
      float bo = b_out[col];
#pragma unroll
      for (int vg = 0; vg < 8; ++vg)
        sL[(vg + hi * 8) * SL_STRIDE + col] = acc[mt][nt][vg] + bo;
    }
    __syncthreads();
    for (int r = w; r < 16; r += 8) {
      float mx = -3.4e38f;
      for (int c = lane; c < V_; c += 32) mx = fmaxf(mx, sL[r * SL_STRIDE + c]);
      for (int off = 16; off > 0; off >>= 1) mx = fmaxf(mx, __shfl_xor(mx, off, 32));
      float s = 0.f;
      for (int c = lane; c < V_; c += 32) s += __expf(sL[r * SL_STRIDE + c] - mx);
      for (int off = 16; off > 0; off >>= 1) s += __shfl_xor(s, off, 32);
      float lse = mx + __logf(s);
      if (lane == 0) {
        int p = p0 + mt * 16 + r;
        if (p < TOTALP) {
          int b   = p / (T_ * U1_);
          int rem = p - b * (T_ * U1_);
          int t   = rem / U1_;
          int u   = rem - t * U1_;
          blank_lp[p] = sL[r * SL_STRIDE + 0] - lse;   // p == (b*T+t)*U1+u
          if (u < U_) {
            int tgt = labels[b * U_ + u];
            label_lp[(size_t)(b * T_ + t) * U_ + u] = sL[r * SL_STRIDE + tgt] - lse;
          }
        }
      }
    }
    __syncthreads();
  }
}

// ---------------------------------------------------------------------------
// Kernel 5: RNN-T forward DP, anti-diagonal wavefront, one block per batch.
// ---------------------------------------------------------------------------
__global__ __launch_bounds__(128)
void k_dp(const float* __restrict__ blank_lp, const float* __restrict__ label_lp,
          const int* __restrict__ t_len, const int* __restrict__ u_len,
          float* __restrict__ total) {
  int b  = blockIdx.x;
  int tl = t_len[b], ul = u_len[b];
  const float* blp = blank_lp + (size_t)b * T_ * U1_;
  const float* llp = label_lp + (size_t)b * T_ * U_;
  __shared__ float alpha[2][U1_];
  int tid = threadIdx.x;

  for (int d = 0; d < T_ + U1_ - 1; ++d) {
    float*       cur  = alpha[d & 1];
    const float* prev = alpha[(d & 1) ^ 1];
    int ulo = d - (T_ - 1); if (ulo < 0) ulo = 0;
    int uhi = (d < U_) ? d : U_;                    // u <= min(d, 120)
    for (int u = ulo + tid; u <= uhi; u += 128) {
      int t = d - u;
      float a;
      if (t == 0 && u == 0) {
        a = 0.f;
      } else if (t == 0) {
        float l = (u - 1 < ul) ? llp[u - 1] : NEGV;
        a = prev[u - 1] + l;
      } else if (u == 0) {
        a = prev[0] + blp[(size_t)(t - 1) * U1_];
      } else {
        float pa = prev[u]     + blp[(size_t)(t - 1) * U1_ + u];
        float l  = (u - 1 < ul) ? llp[(size_t)t * U_ + (u - 1)] : NEGV;
        float pb = prev[u - 1] + l;
        float m  = fmaxf(pa, pb);
        a = m + __logf(__expf(pa - m) + __expf(pb - m));
      }
      cur[u] = a;
      if (t == tl - 1 && u == ul)
        total[b] = a + blp[(size_t)(tl - 1) * U1_ + ul];
    }
    __syncthreads();
  }
}

// ---------------------------------------------------------------------------
// Kernel 6: loss = -mean(total_lp)
// ---------------------------------------------------------------------------
__global__ void k_final(const float* __restrict__ total, float* __restrict__ out) {
  if (threadIdx.x == 0) {
    float s = 0.f;
    for (int b = 0; b < B_; ++b) s += total[b];
    out[0] = -s / (float)B_;
  }
}

// ---------------------------------------------------------------------------
extern "C" void kernel_launch(void* const* d_in, const int* in_sizes, int n_in,
                              void* d_out, int out_size, void* d_ws, size_t ws_size,
                              hipStream_t stream) {
  (void)in_sizes; (void)n_in; (void)out_size; (void)ws_size;
  const float* encoder_out = (const float*)d_in[0];
  const int*   enc_lens    = (const int*)  d_in[1];
  const int*   labels      = (const int*)  d_in[2];
  const int*   lab_lens    = (const int*)  d_in[3];
  const float* emb         = (const float*)d_in[4];
  const float* W_enc       = (const float*)d_in[5];
  const float* W_dec       = (const float*)d_in[6];
  const float* b_joint     = (const float*)d_in[7];
  const float* W_out       = (const float*)d_in[8];
  const float* b_out       = (const float*)d_in[9];

  char* ws = (char*)d_ws;
  size_t off = 0;
  auto alloc = [&](size_t bytes) -> void* {
    void* p = ws + off;
    off = (off + bytes + 255) & ~(size_t)255;
    return p;
  };
  float*  enc_p    = (float*) alloc((size_t)B_ * T_  * H_ * 4);
  float*  dec_p    = (float*) alloc((size_t)B_ * U1_ * H_ * 4);
  __bf16* WT       = (__bf16*)alloc((size_t)V_ * H_ * 2);
  float*  blank_lp = (float*) alloc((size_t)B_ * T_ * U1_ * 4);
  float*  label_lp = (float*) alloc((size_t)B_ * T_ * U_  * 4);
  float*  total    = (float*) alloc((size_t)B_ * 4);

  k_wt   <<<(V_ * H_) / 256, 256, 0, stream>>>(W_out, WT);
  k_encp <<<B_ * T_,        256, 0, stream>>>(encoder_out, W_enc, enc_p);
  k_decp <<<B_ * U1_,       256, 0, stream>>>(labels, emb, W_dec, dec_p);
  k_joint<<<NBLK,           256, SMEM_BYTES, stream>>>(enc_p, dec_p, b_joint, WT,
                                                       b_out, labels, blank_lp,
                                                       label_lp);
  k_dp   <<<B_,             128, 0, stream>>>(blank_lp, label_lp, enc_lens,
                                              lab_lens, total);
  k_final<<<1, 32, 0, stream>>>(total, (float*)d_out);
}
